// ResidualVectorQuantizer_5488968204712
// MI455X (gfx1250) — matrix operand order repaired and checked by reading, not compiled, not run on hardware
//
#include <hip/hip_runtime.h>
#include <hip/hip_bf16.h>

// ResidualVectorQuantizer for MI455X (gfx1250, wave32, WMMA).
// score[m][k] = x_m . c_k - 0.5*||c_k||^2 ; argmax == argmin of L2 distance.
// GEMM via v_wmma_f32_16x16x32_f16; B codebook staged through LDS once per
// workgroup (8 waves share it) with a 2-deep software pipeline.

typedef __attribute__((ext_vector_type(16))) _Float16 v16h;
typedef __attribute__((ext_vector_type(8)))  _Float16 v8h;
typedef __attribute__((ext_vector_type(8)))  float    v8f;

#define WAVES_PER_BLOCK 8
#define TILE_M   16
#define CT       8                    // code tiles per LDS chunk
#define NT       256                  // K/16 (K = 4096)
#define NTP      (NT + 4)             // padded hn row stride (bank-conflict-free)
#define CHUNK_F4 1024                 // CT * 2048B / 16B
#define CHUNK_H  8192                 // CT * 2048B / 2B (halves)

// ---------------------------------------------------------------------------
// Pack f32 [K,64] codebook into the WMMA B layout, in 16B sub-blocks:
//   16B unit u = ((tile*2 + kstep)*2 + sb)*32 + lane ; half h = sb*8 + h0
//   column  N = tile*16 + (lane & 15)
//   dim     d = kstep*32 + sb*16 + (lane >> 4)*8 + h0 (ISA 7.12.2 B layout)
// ---------------------------------------------------------------------------
__global__ __launch_bounds__(256) void rvq_pack_b(const float* __restrict__ cb,
                                                  _Float16* __restrict__ packed,
                                                  int K) {
  int t = blockIdx.x * blockDim.x + threadIdx.x;
  if (t >= K * 64) return;
  int h0   = t & 7;
  int u    = t >> 3;
  int lane = u & 31;
  int sb   = (u >> 5) & 1;
  int s    = (u >> 6) & 1;
  int j    = u >> 7;
  int h    = sb * 8 + h0;
  int code = j * 16 + (lane & 15);
  int d    = s * 32 + sb * 16 + (lane >> 4) * 8 + h0;
  packed[t] = (_Float16)cb[(size_t)code * 64 + d];
}

__global__ __launch_bounds__(256) void rvq_half_norms(const float* __restrict__ cb,
                                                      float* __restrict__ hn, int K) {
  int k = blockIdx.x * blockDim.x + threadIdx.x;
  if (k >= K) return;
  float s = 0.f;
  #pragma unroll
  for (int d = 0; d < 64; ++d) { float v = cb[(size_t)k * 64 + d]; s += v * v; }
  hn[k] = 0.5f * s;
}

// A-operand packing (ISA 7.12.2 16-bit A 16x32, wave32):
//   group g = lane>>4, row M = lane&15
//   halves 0..7 -> K = g*8 + h ; halves 8..15 -> K = 16 + g*8 + (h-8)
__device__ inline v16h load_a_half(const float* __restrict__ rowp, int s, int g) {
  v16h a;
  const float* p0 = rowp + s * 32 + g * 8;
  const float* p1 = rowp + s * 32 + 16 + g * 8;
  #pragma unroll
  for (int i = 0; i < 8; ++i) a[i]     = (_Float16)p0[i];
  #pragma unroll
  for (int i = 0; i < 8; ++i) a[i + 8] = (_Float16)p1[i];
  return a;
}

// ---------------------------------------------------------------------------
// One VQ stage. 8 waves/block; each wave owns a 16-row tile but all waves scan
// the same code tiles, so B chunks are staged cooperatively through LDS.
// ---------------------------------------------------------------------------
__global__ __launch_bounds__(256) void rvq_vq_stage(
    const float*    __restrict__ src,       // [N,64] rows to quantize
    const _Float16* __restrict__ packedB,   // packed codebook (B layout)
    const float*    __restrict__ hn,        // [K] 0.5*||c_k||^2
    const float*    __restrict__ cbf,       // [K,64] f32 codebook for gather
    float*          __restrict__ idx_out,   // [N]
    float*          __restrict__ z_out,     // [N,64]
    float*          __restrict__ xcopy_out, // stage1 only
    float*          __restrict__ resid_out, // stage1 only
    const float*    __restrict__ zprev,     // stage2 only
    float*          __restrict__ sum_out,   // stage2 only
    int K) {
  __shared__ __attribute__((aligned(16))) _Float16 sB[2 * CHUNK_H]; // 32 KB, 2 buffers
  __shared__ __attribute__((aligned(16))) float    sHn[16 * NTP];   // padded hn
  __shared__ int lds_idx[WAVES_PER_BLOCK][16];

  const int tid  = threadIdx.x;
  const int lane = tid & 31;
  const int wave = tid >> 5;
  const int g    = lane >> 4;
  const int n    = lane & 15;
  const int row_base = (blockIdx.x * WAVES_PER_BLOCK + wave) * TILE_M;

  const int ntiles  = K >> 4;          // 256
  const int nchunks = ntiles / CT;     // 32

  // ---- A tile: 16 rows x 64 dims, f32 -> f16, register resident ----
  const float* rowp = src + (size_t)(row_base + n) * 64;
  v16h a0 = load_a_half(rowp, 0, g);
  v16h a1 = load_a_half(rowp, 1, g);

  // ---- cooperative fills: hn -> LDS (padded), B chunk0 -> LDS, chunk1 -> regs
  for (int k = tid; k < K; k += 256)
    sHn[(k & 15) * NTP + (k >> 4)] = hn[k];

  const float4* gB4 = (const float4*)packedB;
  float4 stage[4];
  #pragma unroll
  for (int i = 0; i < 4; ++i) stage[i] = gB4[(size_t)0 * CHUNK_F4 + i * 256 + tid];
  {
    float4* dst = (float4*)&sB[0];
    #pragma unroll
    for (int i = 0; i < 4; ++i) dst[i * 256 + tid] = stage[i];
  }
  if (nchunks > 1) {
    #pragma unroll
    for (int i = 0; i < 4; ++i) stage[i] = gB4[(size_t)1 * CHUNK_F4 + i * 256 + tid];
  }
  __syncthreads();

  // ---- scan all codes ----
  float best_v[8];
  int   best_i[8];
  #pragma unroll
  for (int r = 0; r < 8; ++r) { best_v[r] = -3.0e38f; best_i[r] = 0; }

  int code = n;
  for (int ch = 0; ch < nchunks; ++ch) {
    // hn for this chunk's 8 tiles: two conflict-free ds_load_b128
    const float4* hn4 = (const float4*)&sHn[n * NTP + ch * CT];
    float4 h0 = hn4[0], h1 = hn4[1];
    float hnv[8] = {h0.x, h0.y, h0.z, h0.w, h1.x, h1.y, h1.z, h1.w};

    const v8h* cb8 = (const v8h*)&sB[(ch & 1) ? CHUNK_H : 0];

    // HBM prefetch a few chunks out (global_prefetch_b8)
    if (ch + 3 < nchunks)
      __builtin_prefetch((const void*)&gB4[(size_t)(ch + 3) * CHUNK_F4 + tid], 0, 1);

    #pragma unroll
    for (int tt = 0; tt < CT; ++tt) {
      v8f c;
      #pragma unroll
      for (int r = 0; r < 8; ++r) c[r] = -hnv[tt];   // fold -0.5||c||^2

      v8h lo0 = cb8[((tt * 2 + 0) * 2 + 0) * 32 + lane];
      v8h hi0 = cb8[((tt * 2 + 0) * 2 + 1) * 32 + lane];
      v16h b0 = __builtin_shufflevector(lo0, hi0, 0, 1, 2, 3, 4, 5, 6, 7,
                                        8, 9, 10, 11, 12, 13, 14, 15);
      c = __builtin_amdgcn_wmma_f32_16x16x32_f16(false, a0, false, b0,
                                                 (short)0, c, false, false);
      v8h lo1 = cb8[((tt * 2 + 1) * 2 + 0) * 32 + lane];
      v8h hi1 = cb8[((tt * 2 + 1) * 2 + 1) * 32 + lane];
      v16h b1 = __builtin_shufflevector(lo1, hi1, 0, 1, 2, 3, 4, 5, 6, 7,
                                        8, 9, 10, 11, 12, 13, 14, 15);
      c = __builtin_amdgcn_wmma_f32_16x16x32_f16(false, a1, false, b1,
                                                 (short)0, c, false, false);

      // C layout: reg r, lane -> row M = g*8+r, code N = n
      #pragma unroll
      for (int r = 0; r < 8; ++r) {
        bool gt = c[r] > best_v[r];                  // strict > keeps lowest index
        best_v[r] = gt ? c[r] : best_v[r];
        best_i[r] = gt ? code : best_i[r];
      }
      code += 16;
    }

    // rotate the software pipeline: regs(ch+1) -> LDS, issue globals for ch+2
    if (ch + 1 < nchunks) {
      float4* dst = (float4*)&sB[((ch + 1) & 1) ? CHUNK_H : 0];
      #pragma unroll
      for (int i = 0; i < 4; ++i) dst[i * 256 + tid] = stage[i];
      if (ch + 2 < nchunks) {
        #pragma unroll
        for (int i = 0; i < 4; ++i)
          stage[i] = gB4[(size_t)(ch + 2) * CHUNK_F4 + i * 256 + tid];
      }
    }
    __syncthreads();
  }

  // ---- argmax across the 16 N-lanes of each group ----
  #pragma unroll
  for (int m = 1; m <= 8; m <<= 1) {
    #pragma unroll
    for (int r = 0; r < 8; ++r) {
      float v2 = __shfl_xor(best_v[r], m, 32);
      int   i2 = __shfl_xor(best_i[r], m, 32);
      bool take = (v2 > best_v[r]) || ((v2 == best_v[r]) && (i2 < best_i[r]));
      best_v[r] = take ? v2 : best_v[r];
      best_i[r] = take ? i2 : best_i[r];
    }
  }
  if (n == 0) {
    #pragma unroll
    for (int r = 0; r < 8; ++r) lds_idx[wave][g * 8 + r] = best_i[r];
  }
  __syncthreads();

  // ---- writeback: lane covers (row = n, dims [g*32, g*32+32)) ----
  {
    const int row = n;
    const int d0  = g * 32;
    const int idx = lds_idx[wave][row];
    const size_t ro = (size_t)(row_base + row) * 64 + d0;
    const float4* zc = (const float4*)(cbf + (size_t)idx * 64 + d0);
    const float4* xs = (const float4*)(src + ro);
    float4* zo = (float4*)(z_out + ro);
    float4* xo = xcopy_out ? (float4*)(xcopy_out + ro) : nullptr;
    float4* rv = resid_out ? (float4*)(resid_out + ro) : nullptr;
    float4* so = sum_out   ? (float4*)(sum_out   + ro) : nullptr;
    const float4* zp = zprev ? (const float4*)(zprev + ro) : nullptr;
    #pragma unroll
    for (int q = 0; q < 8; ++q) {
      float4 cv = zc[q];
      float4 xv = xs[q];
      zo[q] = cv;
      if (xo) xo[q] = xv;
      if (rv) rv[q] = make_float4(xv.x - cv.x, xv.y - cv.y, xv.z - cv.z, xv.w - cv.w);
      if (so) { float4 z0 = zp[q];
                so[q] = make_float4(z0.x + cv.x, z0.y + cv.y, z0.z + cv.z, z0.w + cv.w); }
    }
    if (g == 0) idx_out[row_base + row] = (float)idx;
  }
}

// ---------------------------------------------------------------------------
// Host launcher. Output tuple, flattened in return order:
//   [0] z+inner_z (N*D)  [1] z (N*D)  [2] x (N*D)  [3] indices (N)
//   [4] inner_z (N*D)    [5] residual (N*D)        [6] inner_indices (N)
// ---------------------------------------------------------------------------
extern "C" void kernel_launch(void* const* d_in, const int* in_sizes, int n_in,
                              void* d_out, int out_size, void* d_ws, size_t ws_size,
                              hipStream_t stream) {
  const float* x   = (const float*)d_in[0];
  const float* cb  = (const float*)d_in[1];
  const float* icb = (const float*)d_in[2];

  const int D = 64;
  const int N = in_sizes[0] / D;
  const int K = in_sizes[1] / D;

  float* out      = (float*)d_out;
  float* out_sum  = out;
  float* out_z    = out + (size_t)1 * N * D;
  float* out_x    = out + (size_t)2 * N * D;
  float* out_idx  = out + (size_t)3 * N * D;
  float* out_iz   = out + (size_t)3 * N * D + N;
  float* out_res  = out + (size_t)4 * N * D + N;
  float* out_iidx = out + (size_t)5 * N * D + N;

  // workspace: packed f16 codebooks + half-norms
  char* ws = (char*)d_ws;
  _Float16* pB   = (_Float16*)(ws);
  _Float16* piB  = (_Float16*)(ws + (size_t)K * D * 2);
  float*    hnB  = (float*)   (ws + (size_t)K * D * 4);
  float*    hniB = (float*)   (ws + (size_t)K * D * 4 + (size_t)K * 4);

  const int packElems = K * D;
  rvq_pack_b<<<(packElems + 255) / 256, 256, 0, stream>>>(cb,  pB,  K);
  rvq_pack_b<<<(packElems + 255) / 256, 256, 0, stream>>>(icb, piB, K);
  rvq_half_norms<<<(K + 255) / 256, 256, 0, stream>>>(cb,  hnB,  K);
  rvq_half_norms<<<(K + 255) / 256, 256, 0, stream>>>(icb, hniB, K);

  const int blocks = N / (WAVES_PER_BLOCK * TILE_M);   // 512 blocks of 8 waves

  // Stage 1: outer VQ on x -> indices, z, x copy, residual
  rvq_vq_stage<<<blocks, 256, 0, stream>>>(
      x, pB, hnB, cb,
      out_idx, out_z, out_x, out_res, nullptr, nullptr, K);

  // Stage 2: inner VQ on residual -> inner_indices, inner_z, z+inner_z
  rvq_vq_stage<<<blocks, 256, 0, stream>>>(
      out_res, piB, hniB, icb,
      out_iidx, out_iz, nullptr, nullptr, out_z, out_sum, K);
}
